// GPT_15015205667179
// MI455X (gfx1250) — compile-verified, matrix operand order checked by reading, not run on hardware
//
#include <hip/hip_runtime.h>
#include <hip/hip_bf16.h>

typedef __attribute__((ext_vector_type(16))) __bf16 v16bf;
typedef __attribute__((ext_vector_type(8)))  __bf16 bf16x8;
typedef __attribute__((ext_vector_type(8)))  float  v8f;
typedef __attribute__((ext_vector_type(4)))  float  f32x4;

#define Vv   32000
#define Dm   512
#define Hh   8
#define Ll   6
#define DFF  2048
#define Bb   2
#define Ss   1024
#define DK   64
#define BS   (Bb*Ss)

// ---------------------------------------------------------------------------
// Embedding: h[b,s,d] = tok_emb[x[b,s], d] + pos_emb[s, d]
// ---------------------------------------------------------------------------
__global__ __launch_bounds__(256) void embed_kernel(
    const int* __restrict__ x, const float* __restrict__ tok,
    const float* __restrict__ pos, float* __restrict__ h)
{
  int idx = blockIdx.x * 256 + threadIdx.x;          // over B*S*D
  int d = idx & (Dm - 1);
  int t = idx / Dm;                                  // b*S + s
  int s = t & (Ss - 1);
  int tokid = x[t];
  h[idx] = tok[(size_t)tokid * Dm + d] + pos[(size_t)s * Dm + d];
}

// ---------------------------------------------------------------------------
// LayerNorm over D=512, one block (128 thr) per token
// ---------------------------------------------------------------------------
__global__ __launch_bounds__(128) void ln_kernel(
    const float* __restrict__ x, const float* __restrict__ g,
    const float* __restrict__ b, float* __restrict__ y)
{
  const int row = blockIdx.x;
  const float* xr = x + (size_t)row * Dm;
  float* yr = y + (size_t)row * Dm;
  __shared__ float red[4];

  float s = 0.f;
  for (int i = threadIdx.x; i < Dm; i += 128) s += xr[i];
  #pragma unroll
  for (int off = 16; off > 0; off >>= 1) s += __shfl_xor(s, off, 32);
  if ((threadIdx.x & 31) == 0) red[threadIdx.x >> 5] = s;
  __syncthreads();
  float mean = (red[0] + red[1] + red[2] + red[3]) * (1.0f / Dm);
  __syncthreads();

  float vs = 0.f;
  for (int i = threadIdx.x; i < Dm; i += 128) { float d = xr[i] - mean; vs += d * d; }
  #pragma unroll
  for (int off = 16; off > 0; off >>= 1) vs += __shfl_xor(vs, off, 32);
  if ((threadIdx.x & 31) == 0) red[threadIdx.x >> 5] = vs;
  __syncthreads();
  float var = (red[0] + red[1] + red[2] + red[3]) * (1.0f / Dm);
  float rstd = rsqrtf(var + 1e-5f);
  for (int i = threadIdx.x; i < Dm; i += 128)
    yr[i] = (xr[i] - mean) * rstd * g[i] + b[i];
}

// ---------------------------------------------------------------------------
// GEMM: C[M,N] = A[M,K] * B(+transB) + bias (+residual)(+relu)
// fp32 in/out, bf16 WMMA. Block tile 128x64, 8 waves (256 thr), K-step 32.
// Software-pipelined: double-buffered LDS, next tile's global_load_b128
// clause issued before the current tile's WMMAs, one barrier per K-step.
// All five fragment loads (1xA + 4xB) are hoisted into registers ahead of the
// WMMA sequence so the ds clause issues once and the 4 WMMAs run back-to-back.
// LDS tiles live in WMMA *fragment* layout (ISA 7.12.2):
//     lane  = idx16 + 16*( (k & 15) >= 8 )     idx16 = row(A)/col(B) % 16
//     slot  = (k < 16) ? (k & 7) : 8 + (k & 7)
// Requires M%128==0, N%64==0, K%32==0 (true for all call sites here).
// ---------------------------------------------------------------------------
__global__ __launch_bounds__(256) void gemm_bf16w(
    const float* __restrict__ A, const float* __restrict__ B,
    const float* __restrict__ bias, const float* __restrict__ residual,
    float* __restrict__ C, int M, int N, int K, int transB, int relu)
{
  __shared__ alignas(32) __bf16 Af[2][8][32][16];   // [buf][row strip][lane][slot]
  __shared__ alignas(32) __bf16 Bf[2][4][32][16];   // [buf][col chunk][lane][slot]

  const int m0 = blockIdx.y * 128;
  const int n0 = blockIdx.x * 64;
  const int tid   = threadIdx.x;
  const int wave  = tid >> 5;
  const int lane  = tid & 31;
  const int l16   = lane & 15;
  const int lhalf = lane >> 4;

  // staging tasks: A: 128 rows x 2 k-groups -> all 256 threads
  //                B:  64 cols x 2 k-groups -> first 128 threads
  const int ar = tid & 127, ag = tid >> 7;
  const int bc = tid & 63,  bg = (tid >> 6) & 1;
  const bool doB = tid < 128;

  const float* aptr = A + (size_t)(m0 + ar) * K + ag * 16;

  f32x4 Ar[4];
  float Br[16];

  auto loadA = [&](int kb) {
    const f32x4* p = (const f32x4*)(aptr + kb);
    Ar[0] = p[0]; Ar[1] = p[1]; Ar[2] = p[2]; Ar[3] = p[3];
  };
  auto loadB = [&](int kb) {
    if (!doB) return;
    if (transB) {
      const f32x4* p = (const f32x4*)(B + (size_t)(n0 + bc) * K + kb + bg * 16);
      f32x4 b0 = p[0], b1 = p[1], b2 = p[2], b3 = p[3];
      #pragma unroll
      for (int j = 0; j < 4; ++j) {
        Br[j] = b0[j]; Br[4 + j] = b1[j]; Br[8 + j] = b2[j]; Br[12 + j] = b3[j];
      }
    } else {
      #pragma unroll
      for (int j = 0; j < 16; ++j)
        Br[j] = B[(size_t)(kb + bg * 16 + j) * N + n0 + bc];
    }
  };
  auto stage = [&](int buf) {
    bf16x8 lo, hi;
    #pragma unroll
    for (int j = 0; j < 4; ++j) {
      lo[j] = (__bf16)Ar[0][j]; lo[4 + j] = (__bf16)Ar[1][j];
      hi[j] = (__bf16)Ar[2][j]; hi[4 + j] = (__bf16)Ar[3][j];
    }
    *(bf16x8*)&Af[buf][ar >> 4][(ar & 15)     ][ag * 8] = lo;  // k%16 < 8
    *(bf16x8*)&Af[buf][ar >> 4][(ar & 15) + 16][ag * 8] = hi;  // k%16 >= 8
    if (doB) {
      bf16x8 blo, bhi;
      #pragma unroll
      for (int j = 0; j < 8; ++j) { blo[j] = (__bf16)Br[j]; bhi[j] = (__bf16)Br[8 + j]; }
      *(bf16x8*)&Bf[buf][bc >> 4][(bc & 15)     ][bg * 8] = blo;
      *(bf16x8*)&Bf[buf][bc >> 4][(bc & 15) + 16][bg * 8] = bhi;
    }
  };

  v8f acc[4];
  const v8f z = {0.f,0.f,0.f,0.f,0.f,0.f,0.f,0.f};
  #pragma unroll
  for (int c = 0; c < 4; ++c) acc[c] = z;

  // prologue: tile 0 -> buf 0
  loadA(0); loadB(0); stage(0);
  __syncthreads();

  int cur = 0;
  for (int kb = 0; kb < K; kb += 32) {
    const bool more = (kb + 32) < K;
    if (more) { loadA(kb + 32); loadB(kb + 32); }   // in flight under WMMAs

    // hoist ALL fragment loads -> one ds clause, WMMAs back-to-back
    const v16bf af = *(const v16bf*)&Af[cur][wave][lane][0];
    v16bf bfv[4];
    #pragma unroll
    for (int c = 0; c < 4; ++c)
      bfv[c] = *(const v16bf*)&Bf[cur][c][lane][0];
    #pragma unroll
    for (int c = 0; c < 4; ++c)
      acc[c] = __builtin_amdgcn_wmma_f32_16x16x32_bf16(
          false, af, false, bfv[c], (short)0, acc[c], false, false);

    if (more) {
      stage(cur ^ 1);
      __syncthreads();
      cur ^= 1;
    }
  }

  #pragma unroll
  for (int c = 0; c < 4; ++c) {
    const int col = n0 + c * 16 + l16;
    #pragma unroll
    for (int v = 0; v < 8; ++v) {
      const int row = m0 + wave * 16 + v + lhalf * 8;
      float val = acc[c][v];
      if (bias)     val += bias[col];
      if (residual) val += residual[(size_t)row * N + col];
      if (relu)     val = fmaxf(val, 0.0f);
      C[(size_t)row * N + col] = val;
    }
  }
}

// ---------------------------------------------------------------------------
// Flash-style causal attention, one wave per (b*H, 16-query tile).
// S^T = K_tile(16xDK) x Q^T(DKx16) via WMMA. The C-layout of S^T held by a
// lane (query = lane%16, keys = 8*(lane/16)+v) is IDENTICAL to the A-fragment
// layout the P(16q x 32k) x V(32k x 64) WMMA needs from that lane -> P never
// leaves registers. Softmax bookkeeping is lane-local with a lane^16 pair
// combine; per-row broadcast uses __shfl (no LDS at all).
// ---------------------------------------------------------------------------
__global__ __launch_bounds__(32) void attn_kernel(
    const float* __restrict__ Q, const float* __restrict__ K,
    const float* __restrict__ V, float* __restrict__ O)
{
  const int t  = blockIdx.x;            // query tile: rows q0..q0+15
  const int bh = blockIdx.y;            // b*H + h
  const int b  = bh >> 3;
  const int hh = bh & 7;
  const int q0 = t * 16;

  const int lane  = threadIdx.x;
  const int lhalf = lane >> 4;
  const int l16   = lane & 15;
  const int k0    = lhalf * 8;

  const size_t base = ((size_t)b * Ss) * Dm + hh * DK;

  // Q fragments (B-operand of S^T): column = query l16, K-dim = dk
  v16bf qf[2];
  #pragma unroll
  for (int c = 0; c < 2; ++c) {
    const float* qp = Q + base + (size_t)(q0 + l16) * Dm + c * 32;
    f32x4 a0 = *(const f32x4*)(qp + k0);
    f32x4 a1 = *(const f32x4*)(qp + k0 + 4);
    f32x4 a2 = *(const f32x4*)(qp + 16 + k0);
    f32x4 a3 = *(const f32x4*)(qp + 16 + k0 + 4);
    #pragma unroll
    for (int j = 0; j < 4; ++j) {
      qf[c][j]      = (__bf16)a0[j];
      qf[c][4 + j]  = (__bf16)a1[j];
      qf[c][8 + j]  = (__bf16)a2[j];
      qf[c][12 + j] = (__bf16)a3[j];
    }
  }

  v8f of[4];
  const v8f z = {0.f,0.f,0.f,0.f,0.f,0.f,0.f,0.f};
  #pragma unroll
  for (int c = 0; c < 4; ++c) of[c] = z;

  float mloc = -3.0e38f, lloc = 0.f;    // state for query l16 (dup across halves)
  const float scale = 0.125f;           // 1/sqrt(64)
  const int qglob = q0 + l16;
  const int nsub = t + 1;               // valid 16-key subtiles (causal)
  const int nch  = (nsub + 1) >> 1;     // 32-key chunks

  for (int kb = 0; kb < nch; ++kb) {
    float sv[2][8];
    int   valid[2];
    float cmax = -3.0e38f;

    #pragma unroll
    for (int sub = 0; sub < 2; ++sub) {
      const int sidx = kb * 2 + sub;
      valid[sub] = (sidx <= t);
      if (!valid[sub]) continue;
      const int kbase = sidx * 16;

      // load both K fragments first, then both WMMAs back-to-back
      v16bf kf[2];
      #pragma unroll
      for (int c = 0; c < 2; ++c) {
        const float* kp = K + base + (size_t)(kbase + l16) * Dm + c * 32;
        f32x4 a0 = *(const f32x4*)(kp + k0);
        f32x4 a1 = *(const f32x4*)(kp + k0 + 4);
        f32x4 a2 = *(const f32x4*)(kp + 16 + k0);
        f32x4 a3 = *(const f32x4*)(kp + 16 + k0 + 4);
        #pragma unroll
        for (int j = 0; j < 4; ++j) {
          kf[c][j]      = (__bf16)a0[j];
          kf[c][4 + j]  = (__bf16)a1[j];
          kf[c][8 + j]  = (__bf16)a2[j];
          kf[c][12 + j] = (__bf16)a3[j];
        }
      }
      v8f st = z;
      #pragma unroll
      for (int c = 0; c < 2; ++c)
        st = __builtin_amdgcn_wmma_f32_16x16x32_bf16(
            false, kf[c], false, qf[c], (short)0, st, false, false);

      // rows = keys (v + 8*lhalf), cols = queries (l16)
      #pragma unroll
      for (int v = 0; v < 8; ++v) {
        const int kg = kbase + v + k0;
        float x = st[v] * scale;
        if (kg > qglob) x = -3.0e38f;
        sv[sub][v] = x;
        cmax = fmaxf(cmax, x);
      }
    }

    cmax = fmaxf(cmax, __shfl_xor(cmax, 16, 32));
    const float mnew = fmaxf(mloc, cmax);
    const float r = __expf(mloc - mnew);
    float psum = 0.f;

    // Build the P A-fragment IN REGISTERS: pf[sub*8+v] is exactly
    // A[query=l16][k = sub*16 + k0 + v], produced by this very lane.
    v16bf pf;
    #pragma unroll
    for (int sub = 0; sub < 2; ++sub) {
      #pragma unroll
      for (int v = 0; v < 8; ++v) {
        float p = valid[sub] ? __expf(sv[sub][v] - mnew) : 0.f;
        psum += p;
        pf[sub * 8 + v] = (__bf16)p;
      }
    }
    psum += __shfl_xor(psum, 16, 32);
    lloc = lloc * r + psum;
    mloc = mnew;

    // rescale accumulators: factor for row q = v + 8*lhalf lives in lane q
    #pragma unroll
    for (int v = 0; v < 8; ++v) {
      const float fv = __shfl(r, k0 + v, 32);
      #pragma unroll
      for (int c = 0; c < 4; ++c) of[c][v] *= fv;
    }

    // O += P(16q x 32k) x V(32k x 64dk): preload all 4 V fragments,
    // then 4 WMMAs back-to-back.
    const int kbase32 = kb * 32;
    v16bf vf[4];
    #pragma unroll
    for (int c = 0; c < 4; ++c) {
      const float* vp = V + base + c * 16 + l16;
      #pragma unroll
      for (int j = 0; j < 8; ++j) {
        vf[c][j]     = (__bf16)vp[(size_t)(kbase32 + k0 + j) * Dm];
        vf[c][8 + j] = (__bf16)vp[(size_t)(kbase32 + 16 + k0 + j) * Dm];
      }
    }
    #pragma unroll
    for (int c = 0; c < 4; ++c)
      of[c] = __builtin_amdgcn_wmma_f32_16x16x32_bf16(
          false, pf, false, vf[c], (short)0, of[c], false, false);
  }

  // normalize: l for row q = v + 8*lhalf lives in lane q
  float lf[8];
  #pragma unroll
  for (int v = 0; v < 8; ++v) lf[v] = __shfl(lloc, k0 + v, 32);
  #pragma unroll
  for (int c = 0; c < 4; ++c) {
    #pragma unroll
    for (int v = 0; v < 8; ++v) {
      const int row = q0 + v + k0;
      O[base + (size_t)row * Dm + c * 16 + l16] = of[c][v] / lf[v];
    }
  }
}

// ---------------------------------------------------------------------------
extern "C" void kernel_launch(void* const* d_in, const int* in_sizes, int n_in,
                              void* d_out, int out_size, void* d_ws, size_t ws_size,
                              hipStream_t stream) {
  const int*   x       = (const int*)  d_in[0];
  const float* tok_emb = (const float*)d_in[1];
  const float* pos_emb = (const float*)d_in[2];
  const float* Wq      = (const float*)d_in[3];
  const float* bq      = (const float*)d_in[4];
  const float* Wk      = (const float*)d_in[5];
  const float* bk      = (const float*)d_in[6];
  const float* Wv      = (const float*)d_in[7];
  const float* bv      = (const float*)d_in[8];
  const float* Wo      = (const float*)d_in[9];
  const float* bo      = (const float*)d_in[10];
  const float* ln1_g   = (const float*)d_in[11];
  const float* ln1_b   = (const float*)d_in[12];
  const float* ln2_g   = (const float*)d_in[13];
  const float* ln2_b   = (const float*)d_in[14];
  const float* W1      = (const float*)d_in[15];
  const float* b1      = (const float*)d_in[16];
  const float* W2      = (const float*)d_in[17];
  const float* b2      = (const float*)d_in[18];
  const float* lnf_g   = (const float*)d_in[19];
  const float* lnf_b   = (const float*)d_in[20];
  float* logits = (float*)d_out;

  // workspace layout (floats)
  float* ws  = (float*)d_ws;
  const size_t szTok = (size_t)BS * Dm;       // 2048*512
  float* h   = ws;
  float* hn  = h   + szTok;
  float* q   = hn  + szTok;
  float* k   = q   + szTok;
  float* v   = k   + szTok;
  float* ctx = v   + szTok;
  float* ffn = ctx + szTok;                   // BS*DFF

  embed_kernel<<<(BS * Dm) / 256, 256, 0, stream>>>(x, tok_emb, pos_emb, h);

  const dim3 gGemmD (Dm  / 64, BS / 128);     // N=512
  const dim3 gGemmF (DFF / 64, BS / 128);     // N=2048
  const dim3 gGemmV (Vv  / 64, BS / 128);     // N=32000
  const dim3 gAttn  (Ss / 16, Bb * Hh);

  for (int l = 0; l < Ll; ++l) {
    const float* Wq_l = Wq + (size_t)l * Dm * Dm;
    const float* Wk_l = Wk + (size_t)l * Dm * Dm;
    const float* Wv_l = Wv + (size_t)l * Dm * Dm;
    const float* Wo_l = Wo + (size_t)l * Dm * Dm;
    const float* W1_l = W1 + (size_t)l * Dm * DFF;
    const float* W2_l = W2 + (size_t)l * DFF * Dm;

    ln_kernel<<<BS, 128, 0, stream>>>(h, ln1_g + l * Dm, ln1_b + l * Dm, hn);
    gemm_bf16w<<<gGemmD, 256, 0, stream>>>(hn, Wq_l, bq + l * Dm, nullptr, q,
                                           BS, Dm, Dm, 0, 0);
    gemm_bf16w<<<gGemmD, 256, 0, stream>>>(hn, Wk_l, bk + l * Dm, nullptr, k,
                                           BS, Dm, Dm, 0, 0);
    gemm_bf16w<<<gGemmD, 256, 0, stream>>>(hn, Wv_l, bv + l * Dm, nullptr, v,
                                           BS, Dm, Dm, 0, 0);
    attn_kernel<<<gAttn, 32, 0, stream>>>(q, k, v, ctx);
    gemm_bf16w<<<gGemmD, 256, 0, stream>>>(ctx, Wo_l, bo + l * Dm, h, h,
                                           BS, Dm, Dm, 0, 0);
    ln_kernel<<<BS, 128, 0, stream>>>(h, ln2_g + l * Dm, ln2_b + l * Dm, hn);
    gemm_bf16w<<<gGemmF, 256, 0, stream>>>(hn, W1_l, b1 + l * DFF, nullptr, ffn,
                                           BS, DFF, Dm, 0, 1);
    gemm_bf16w<<<gGemmD, 256, 0, stream>>>(ffn, W2_l, b2 + l * Dm, h, h,
                                           BS, Dm, DFF, 0, 0);
  }

  ln_kernel<<<BS, 128, 0, stream>>>(h, lnf_g, lnf_b, hn);
  gemm_bf16w<<<gGemmV, 256, 0, stream>>>(hn, tok_emb, nullptr, nullptr, logits,
                                         BS, Vv, Dm, 1, 0);
}
